// SphericalTokenLinear_35545149341995
// MI455X (gfx1250) — matrix operand order repaired and checked by reading, not compiled, not run on hardware
//
#include <hip/hip_runtime.h>
#include <hip/hip_bf16.h>
#include <math.h>

// ---------------------------------------------------------------------------
// SphericalTokenLinear fused kernel for gfx1250 (MI455X)
//   raw  = x @ k_scaled^T          (k_scaled = k/max(||k||,eps) * sigmoid(temp))
//   w    = gelu( sqrt(T) * raw / ||raw_row|| )    (x-normalization cancels)
//   out  = w @ V
// Both GEMMs run on v_wmma_f32_16x16x32_bf16 with fp32 accumulation.
// 2x2 register tiling: each inner iteration does 4 WMMAs from 2 A-frags (LDS)
// and 2 B-frags (global/L2).  Row sum-of-squares folded into GEMM-1 epilogue.
// ---------------------------------------------------------------------------

typedef __attribute__((ext_vector_type(16))) __bf16 v16bf;
typedef __attribute__((ext_vector_type(4)))  __bf16 v4bf;
typedef __attribute__((ext_vector_type(8)))  float  v8f;
typedef __attribute__((ext_vector_type(4)))  float  v4f;
typedef __attribute__((ext_vector_type(4)))  unsigned int v4u;

#define D_IN   1024
#define D_OUT  1024
#define T_TOK  1024
#define BM     32            // rows per workgroup
#define XP     1040          // x-slab pitch   (bf16 elems, 16B-aligned rows + pad)
#define RP     1040          // raw/weights pitch

struct B16x16 { v4u lo, hi; };

__device__ __forceinline__ v16bf frag16(v4u lo, v4u hi) {
  B16x16 t; t.lo = lo; t.hi = hi;
  return __builtin_bit_cast(v16bf, t);
}

// A-fragment (16x32 bf16, row-major source):
// lane l (r=l&15, h=l>>4) holds row r; elems 0..7 = K[h*8 .. h*8+7],
// elems 8..15 = K[16+h*8 .. 16+h*8+7]   (CDNA5 ISA 16-bit A layout)
__device__ __forceinline__ v16bf load_fragA(const __bf16* p) {
  return frag16(*(const v4u*)p, *(const v4u*)(p + 16));
}

// B-fragment (32x16 bf16, B column n sourced from row n of a row-major matrix):
// lane l (n=l&15, h=l>>4) holds col n; elems 0..7 = K[h*8 .. h*8+7],
// elems 8..15 = K[16+h*8 .. 16+h*8+7]
__device__ __forceinline__ v16bf load_fragB(const __bf16* p) {
  return frag16(*(const v4u*)p, *(const v4u*)(p + 16));
}

__device__ __forceinline__ v8f wmma_bf16(v16bf a, v16bf b, v8f c) {
  return __builtin_amdgcn_wmma_f32_16x16x32_bf16(false, a, false, b,
                                                 (short)0, c, false, false);
}

// ---------------------------------------------------------------------------
// Prep 1: k_scaled[t][d] = key[t][d] * sigmoid(temp[t]) / max(||key[t]||, eps)
// ---------------------------------------------------------------------------
__global__ __launch_bounds__(256) void prep_keys_kernel(
    const float* __restrict__ key, const float* __restrict__ temp,
    __bf16* __restrict__ ks) {
  __shared__ float red[256];
  const int t = blockIdx.x;
  const float* row = key + (size_t)t * D_IN;
  float s = 0.f;
  for (int i = threadIdx.x; i < D_IN; i += 256) { float v = row[i]; s += v * v; }
  red[threadIdx.x] = s;
  __syncthreads();
  for (int off = 128; off > 0; off >>= 1) {
    if (threadIdx.x < off) red[threadIdx.x] += red[threadIdx.x + off];
    __syncthreads();
  }
  const float norm = sqrtf(red[0]);
  const float sig  = 1.0f / (1.0f + expf(-temp[t]));
  const float sc   = sig / fmaxf(norm, 1e-12f);
  for (int i = threadIdx.x; i < D_IN; i += 256)
    ks[(size_t)t * D_IN + i] = (__bf16)(row[i] * sc);
}

// ---------------------------------------------------------------------------
// Prep 2: vt[o][t] = (bf16) value[t][o]   (coalesced tiled transpose)
// ---------------------------------------------------------------------------
__global__ __launch_bounds__(256) void prep_vt_kernel(
    const float* __restrict__ val, __bf16* __restrict__ vt) {
  __shared__ float tile[16][17];
  const int tb = blockIdx.x * 16;   // token base
  const int ob = blockIdx.y * 16;   // out-dim base
  tile[threadIdx.y][threadIdx.x] =
      val[(size_t)(tb + threadIdx.y) * D_OUT + ob + threadIdx.x];
  __syncthreads();
  vt[(size_t)(ob + threadIdx.y) * T_TOK + tb + threadIdx.x] =
      (__bf16)tile[threadIdx.x][threadIdx.y];
}

// ---------------------------------------------------------------------------
// Fused main kernel: 256 threads = 8 waves; block owns BM=32 rows;
// wave w owns columns [w*128, w*128+128).
// ---------------------------------------------------------------------------
__global__ __launch_bounds__(256) void fused_stl_kernel(
    const float* __restrict__ x, const __bf16* __restrict__ ks,
    const __bf16* __restrict__ vt, float* __restrict__ out) {
  extern __shared__ char smem[];
  __bf16* xs  = (__bf16*)smem;            // [BM][XP]  x slab, bf16
  __bf16* wls = xs + BM * XP;             // [BM][RP]  raw -> weights, bf16
  float*  ssq = (float*)(wls + BM * RP);  // [BM]      row sum of squares

  const int tid    = threadIdx.x;
  const int lane   = tid & 31;
  const int wave   = tid >> 5;
  const int h      = lane >> 4;           // half-wave index
  const int ln     = lane & 15;
  const int rowB   = blockIdx.x * BM;     // global row base
  const int nBase  = wave * 128;          // this wave's column stripe

  // ---- stage x slab: fp32 global -> bf16 LDS (coalesced, non-temporal) ----
  for (int i = tid; i < BM * D_IN / 4; i += 256) {
    const int col = (i * 4) & (D_IN - 1);
    const int row = (i * 4) >> 10;
    const v4f f = __builtin_nontemporal_load(
        (const v4f*)(x + (size_t)(rowB + row) * D_IN + col));
    v4bf p; p[0] = (__bf16)f[0]; p[1] = (__bf16)f[1];
            p[2] = (__bf16)f[2]; p[3] = (__bf16)f[3];
    *(v4bf*)(xs + row * XP + col) = p;
  }
  if (tid < BM) ssq[tid] = 0.f;
  __syncthreads();

  // ---- phase 1: raw = x @ ks^T,  2x2 register tiles, fold row sum-sq ----
  float sq0[8], sq1[8];
#pragma unroll
  for (int v = 0; v < 8; ++v) { sq0[v] = 0.f; sq1[v] = 0.f; }

  const __bf16* a0p = xs + ln * XP + h * 8;          // M-slab 0 (rows 0..15)
  const __bf16* a1p = a0p + 16 * XP;                 // M-slab 1 (rows 16..31)

  for (int np = 0; np < 4; ++np) {                   // N-tile pairs
    v8f acc00 = {}, acc01 = {}, acc10 = {}, acc11 = {};
    const __bf16* b0p = ks + (size_t)(nBase + np * 32 + ln) * D_IN + h * 8;
    const __bf16* b1p = b0p + (size_t)16 * D_IN;
#pragma unroll 2
    for (int kt = 0; kt < D_IN / 32; ++kt) {
      const v16bf a0 = load_fragA(a0p + kt * 32);
      const v16bf a1 = load_fragA(a1p + kt * 32);
      const v16bf b0 = load_fragB(b0p + kt * 32);
      const v16bf b1 = load_fragB(b1p + kt * 32);
      acc00 = wmma_bf16(a0, b0, acc00);
      acc01 = wmma_bf16(a0, b1, acc01);
      acc10 = wmma_bf16(a1, b0, acc10);
      acc11 = wmma_bf16(a1, b1, acc11);
    }
    const int col0 = nBase + np * 32 + ln;
    const int col1 = col0 + 16;
#pragma unroll
    for (int v = 0; v < 8; ++v) {
      const int r0 = h * 8 + v, r1 = 16 + h * 8 + v;
      wls[r0 * RP + col0] = (__bf16)acc00[v];
      wls[r0 * RP + col1] = (__bf16)acc01[v];
      wls[r1 * RP + col0] = (__bf16)acc10[v];
      wls[r1 * RP + col1] = (__bf16)acc11[v];
      sq0[v] += acc00[v] * acc00[v] + acc01[v] * acc01[v];
      sq1[v] += acc10[v] * acc10[v] + acc11[v] * acc11[v];
    }
  }
#pragma unroll
  for (int v = 0; v < 8; ++v) {
    atomicAdd(&ssq[h * 8 + v],      sq0[v]);
    atomicAdd(&ssq[16 + h * 8 + v], sq1[v]);
  }
  __syncthreads();

  // ---- in-place RMS scale + exact GELU (vectorized over the raw tile) ----
  {
    const int row = tid >> 3;
    const int cb  = (tid & 7) * 128;
    const float scale = 32.0f * rsqrtf(fmaxf(ssq[row], 1e-30f)); // sqrt(1024)=32
    for (int c = 0; c < 128; c += 4) {
      v4bf p = *(v4bf*)(wls + row * RP + cb + c);
#pragma unroll
      for (int j = 0; j < 4; ++j) {
        const float v = (float)p[j] * scale;
        const float g = 0.5f * v * (1.0f + erff(v * 0.70710678118654752f));
        p[j] = (__bf16)g;
      }
      *(v4bf*)(wls + row * RP + cb + c) = p;
    }
  }
  __syncthreads();

  // ---- phase 3: out = W @ V,  2x2 register tiles, non-temporal stores ----
  const __bf16* wa0p = wls + ln * RP + h * 8;
  const __bf16* wa1p = wa0p + 16 * RP;

  for (int np = 0; np < 4; ++np) {
    v8f acc00 = {}, acc01 = {}, acc10 = {}, acc11 = {};
    const __bf16* b0p = vt + (size_t)(nBase + np * 32 + ln) * T_TOK + h * 8;
    const __bf16* b1p = b0p + (size_t)16 * T_TOK;
#pragma unroll 2
    for (int kt = 0; kt < T_TOK / 32; ++kt) {
      const v16bf a0 = load_fragA(wa0p + kt * 32);
      const v16bf a1 = load_fragA(wa1p + kt * 32);
      const v16bf b0 = load_fragB(b0p + kt * 32);
      const v16bf b1 = load_fragB(b1p + kt * 32);
      acc00 = wmma_bf16(a0, b0, acc00);
      acc01 = wmma_bf16(a0, b1, acc01);
      acc10 = wmma_bf16(a1, b0, acc10);
      acc11 = wmma_bf16(a1, b1, acc11);
    }
    const int col0 = nBase + np * 32 + ln;
    const int col1 = col0 + 16;
#pragma unroll
    for (int v = 0; v < 8; ++v) {
      const size_t r0 = (size_t)(rowB + h * 8 + v) * D_OUT;
      const size_t r1 = (size_t)(rowB + 16 + h * 8 + v) * D_OUT;
      __builtin_nontemporal_store(acc00[v], out + r0 + col0);
      __builtin_nontemporal_store(acc01[v], out + r0 + col1);
      __builtin_nontemporal_store(acc10[v], out + r1 + col0);
      __builtin_nontemporal_store(acc11[v], out + r1 + col1);
    }
  }
}

// ---------------------------------------------------------------------------
extern "C" void kernel_launch(void* const* d_in, const int* in_sizes, int n_in,
                              void* d_out, int out_size, void* d_ws, size_t ws_size,
                              hipStream_t stream) {
  (void)in_sizes; (void)n_in; (void)out_size; (void)ws_size;
  const float* x    = (const float*)d_in[0];  // [4,4096,1024]
  const float* key  = (const float*)d_in[1];  // [1024,1024]
  const float* val  = (const float*)d_in[2];  // [1024,1024]
  const float* temp = (const float*)d_in[3];  // [1024,1]
  float* out = (float*)d_out;                 // [4,4096,1024]

  __bf16* ks = (__bf16*)d_ws;                 // 2 MB: k_scaled bf16 [T][D_in]
  __bf16* vt = ks + (size_t)T_TOK * D_IN;     // 2 MB: value^T bf16  [D_out][T]

  prep_keys_kernel<<<T_TOK, 256, 0, stream>>>(key, temp, ks);
  prep_vt_kernel<<<dim3(T_TOK / 16, D_OUT / 16), dim3(16, 16), 0, stream>>>(val, vt);

  const size_t lds_bytes = (size_t)(BM * XP + BM * RP) * sizeof(__bf16)
                         + BM * sizeof(float);
  const int n_rows = 4 * 4096;
  fused_stl_kernel<<<n_rows / BM, 256, lds_bytes, stream>>>(x, ks, vt, out);
}
